// MixtureOfExperts_70188355551257
// MI455X (gfx1250) — matrix operand order repaired and checked by reading, not compiled, not run on hardware
//
#include <hip/hip_runtime.h>
#include <hip/hip_bf16.h>
#include <math.h>

// Problem constants (B=4, T=2048, C=1024, I=2816, E=7)
#define NTOK  8192
#define C_DIM 1024
#define I_DIM 2816
#define E_NUM 7

typedef __attribute__((ext_vector_type(16))) __bf16 v16bf;
typedef __attribute__((ext_vector_type(8)))  __bf16 v8bf;
typedef __attribute__((ext_vector_type(4)))  __bf16 v4bf;
typedef __attribute__((ext_vector_type(8)))  float  v8f;

// ---------------------------------------------------------------------------
// f32 -> bf16 round-to-nearest-even (one-shot convert passes / epilogues only)
// ---------------------------------------------------------------------------
__device__ __forceinline__ __bf16 f2bf(float f) {
  unsigned u = __builtin_bit_cast(unsigned, f);
  unsigned r = u + 0x7FFFu + ((u >> 16) & 1u);
  unsigned short h = (unsigned short)(r >> 16);
  return __builtin_bit_cast(__bf16, h);
}

// One-shot dense f32 -> bf16 conversion (grid-stride, 4 elems/thread/iter)
__global__ void cvt_bf16_kernel(const float* __restrict__ src,
                                __bf16* __restrict__ dst, long n) {
  long i = ((long)blockIdx.x * blockDim.x + threadIdx.x) * 4;
  const long stride = (long)gridDim.x * blockDim.x * 4;
  for (; i + 3 < n; i += stride) {
    const float4 v = *(const float4*)(src + i);
    v4bf o;
    o[0] = f2bf(v.x); o[1] = f2bf(v.y); o[2] = f2bf(v.z); o[3] = f2bf(v.w);
    *(v4bf*)(dst + i) = o;
  }
}

// ---------------------------------------------------------------------------
// WMMA bf16 fragment loaders (CDNA5 16x16x32, wave32) — pure vector loads.
// A 16x32 (MxK): lane&15 = M; lanes 0-15 hold K {0..7, 16..23},
//                lanes 16-31 hold K {8..15, 24..31}.
// B 32x16 (KxN): lane&15 = N (a weight row, row-major along K);
//                lanes 0-15 hold K {0..7, 16..23}, lanes 16-31 hold K {8..15, 24..31}.
// ---------------------------------------------------------------------------
__device__ __forceinline__ v16bf fragA(const __bf16* __restrict__ row,
                                       int k0, int lane) {
  const int kb = k0 + ((lane & 16) ? 8 : 0);
  const v8bf lo = *(const v8bf*)(row + kb);
  const v8bf hi = *(const v8bf*)(row + kb + 16);
  return __builtin_shufflevector(lo, hi, 0, 1, 2, 3, 4, 5, 6, 7,
                                 8, 9, 10, 11, 12, 13, 14, 15);
}

__device__ __forceinline__ v16bf fragB(const __bf16* __restrict__ wrow,
                                       int k0, int lane) {
  const int kb = k0 + ((lane & 16) ? 8 : 0);
  const v8bf lo = *(const v8bf*)(wrow + kb);
  const v8bf hi = *(const v8bf*)(wrow + kb + 16);
  return __builtin_shufflevector(lo, hi, 0, 1, 2, 3, 4, 5, 6, 7,
                                 8, 9, 10, 11, 12, 13, 14, 15);
}

#define WMMA_BF16(A, B, Cacc) \
  __builtin_amdgcn_wmma_f32_16x16x32_bf16(false, (A), false, (B), (short)0, \
                                          (Cacc), false, false)

// ---------------------------------------------------------------------------
// Router: one wave per token (f32 math; negligible cost).
// ---------------------------------------------------------------------------
__global__ void moe_router_kernel(const float* __restrict__ X,
                                  const float* __restrict__ RW,
                                  const float* __restrict__ RB,
                                  float* __restrict__ scale_sh,
                                  float* __restrict__ scale_sw,
                                  int* __restrict__ cnt,
                                  int* __restrict__ list) {
  const int wave = threadIdx.x >> 5;
  const int lane = threadIdx.x & 31;
  const int n = blockIdx.x * 8 + wave;
  if (n >= NTOK) return;
  const float* xr = X + (size_t)n * C_DIM;

  float acc[E_NUM];
#pragma unroll
  for (int e = 0; e < E_NUM; ++e) acc[e] = 0.0f;

  for (int c = lane; c < C_DIM; c += 32) {
    const float xv = xr[c];
#pragma unroll
    for (int e = 0; e < E_NUM; ++e) acc[e] += xv * RW[(size_t)e * C_DIM + c];
  }
#pragma unroll
  for (int e = 0; e < E_NUM; ++e) {
    for (int off = 16; off; off >>= 1) acc[e] += __shfl_xor(acc[e], off, 32);
  }

  if (lane == 0) {
    int best = 0;
    float bp = -1.0f;
#pragma unroll
    for (int e = 0; e < E_NUM; ++e) {
      float lg = acc[e] + RB[e];
      lg = fminf(fmaxf(lg, -50.0f), 50.0f);
      const float p = 1.0f / (1.0f + __expf(-lg));
      if (p > bp) { bp = p; best = e; }   // strict '>' keeps lowest index on tie
    }
    bp = fminf(fmaxf(bp, 1e-8f), 1.0f - 1e-8f);
    float tw = 0.5f + bp;
    tw = fminf(fmaxf(tw + 1e-8f, 0.5f), 2.0f);
    scale_sh[n] = 0.5f / tw;
    scale_sw[n] = bp / tw;
    const int slot = atomicAdd(&cnt[best], 1);
    list[best * NTOK + slot] = n;
  }
}

// ---------------------------------------------------------------------------
// Fused up-projection: act = silu(x@W1^T) * (x@W3^T), bf16 in / bf16 out.
// Block = 4 waves; block tile = 32 tokens x 256 cols; wave tile = 32x64.
// Each B fragment feeds two WMMAs (A row-set 0 and 1): 16 WMMAs per K-step
// for 2 A-loads + 8 B-loads. Grid: (I/256=11, NTOK/32=256, nExperts).
// ---------------------------------------------------------------------------
__global__ void moe_up_kernel(const __bf16* __restrict__ X,
                              const __bf16* __restrict__ W1base,
                              const __bf16* __restrict__ W3base,
                              __bf16* __restrict__ act,
                              const int* __restrict__ tok_list,
                              const int* __restrict__ cnt) {
  const int e = blockIdx.z;
  const __bf16* W1 = W1base + (size_t)e * I_DIM * C_DIM;
  const __bf16* W3 = W3base + (size_t)e * I_DIM * C_DIM;

  const int* list = nullptr;
  int nTok = NTOK;
  if (tok_list) { list = tok_list + (size_t)e * NTOK; nTok = cnt[e]; }

  const int row0 = blockIdx.y * 32;
  if (row0 >= nTok) return;

  const int wave = threadIdx.x >> 5;
  const int lane = threadIdx.x & 31;
  const int col0 = blockIdx.x * 256 + wave * 64;
  const int nlane = lane & 15;

  // Two gathered A rows per lane (row-sets 0 and 1)
  const int r0 = row0 + nlane;
  const int r1 = row0 + 16 + nlane;
  const int tok0 = list ? ((r0 < nTok) ? list[r0] : list[row0]) : r0;
  const int tok1 = list ? ((r1 < nTok) ? list[r1] : list[row0]) : r1;
  const __bf16* arow0 = X + (size_t)tok0 * C_DIM;
  const __bf16* arow1 = X + (size_t)tok1 * C_DIM;

  // Single base pointer per weight matrix; column-tile offset is a constant
  // folded into the load IOFFSET.
  const __bf16* w1b = W1 + (size_t)(col0 + nlane) * C_DIM;
  const __bf16* w3b = W3 + (size_t)(col0 + nlane) * C_DIM;

  const v8f vzero = {0.f, 0.f, 0.f, 0.f, 0.f, 0.f, 0.f, 0.f};
  v8f acc1a[4], acc1b[4], acc3a[4], acc3b[4];
#pragma unroll
  for (int t = 0; t < 4; ++t) {
    acc1a[t] = vzero; acc1b[t] = vzero;
    acc3a[t] = vzero; acc3b[t] = vzero;
  }

#pragma unroll 1
  for (int k0 = 0; k0 < C_DIM; k0 += 32) {
    const v16bf a0 = fragA(arow0, k0, lane);
    const v16bf a1 = fragA(arow1, k0, lane);
#pragma unroll
    for (int t = 0; t < 4; ++t) {
      const v16bf b1 = fragB(w1b + (size_t)t * 16 * C_DIM, k0, lane);
      acc1a[t] = WMMA_BF16(a0, b1, acc1a[t]);
      acc1b[t] = WMMA_BF16(a1, b1, acc1b[t]);
      const v16bf b3 = fragB(w3b + (size_t)t * 16 * C_DIM, k0, lane);
      acc3a[t] = WMMA_BF16(a0, b3, acc3a[t]);
      acc3b[t] = WMMA_BF16(a1, b3, acc3b[t]);
    }
  }

  // Epilogue: SiLU(h1) * h3 -> bf16 activation, scattered by token index.
#pragma unroll
  for (int s = 0; s < 2; ++s) {
    const v8f* h1s = s ? acc1b : acc1a;
    const v8f* h3s = s ? acc3b : acc3a;
#pragma unroll
    for (int g = 0; g < 8; ++g) {
      const int rowIdx = row0 + s * 16 + g + ((lane & 16) ? 8 : 0);
      if (rowIdx >= nTok) continue;
      const int tok = list ? list[rowIdx] : rowIdx;
      __bf16* orow = act + (size_t)tok * I_DIM + col0 + nlane;
#pragma unroll
      for (int t = 0; t < 4; ++t) {
        const float h1 = h1s[t][g];
        const float h3 = h3s[t][g];
        const float sg = h1 / (1.0f + __expf(-h1));
        orow[t * 16] = f2bf(sg * h3);
      }
    }
  }
}

// ---------------------------------------------------------------------------
// Down-projection: y = act @ W2^T (W2 is [C, I]); out[tok] (=) or (+=) scale*y.
// Wave tile 32x64. Grid: (C/256=4, NTOK/32=256, nExperts).
// ---------------------------------------------------------------------------
__global__ void moe_down_kernel(const __bf16* __restrict__ act,
                                const __bf16* __restrict__ W2base,
                                float* __restrict__ out,
                                const int* __restrict__ tok_list,
                                const int* __restrict__ cnt,
                                const float* __restrict__ scale,
                                int accumulate) {
  const int e = blockIdx.z;
  const __bf16* W2 = W2base + (size_t)e * C_DIM * I_DIM;

  const int* list = nullptr;
  int nTok = NTOK;
  if (tok_list) { list = tok_list + (size_t)e * NTOK; nTok = cnt[e]; }

  const int row0 = blockIdx.y * 32;
  if (row0 >= nTok) return;

  const int wave = threadIdx.x >> 5;
  const int lane = threadIdx.x & 31;
  const int col0 = blockIdx.x * 256 + wave * 64;
  const int nlane = lane & 15;

  const int r0 = row0 + nlane;
  const int r1 = row0 + 16 + nlane;
  const int tok0 = list ? ((r0 < nTok) ? list[r0] : list[row0]) : r0;
  const int tok1 = list ? ((r1 < nTok) ? list[r1] : list[row0]) : r1;
  const __bf16* arow0 = act + (size_t)tok0 * I_DIM;
  const __bf16* arow1 = act + (size_t)tok1 * I_DIM;

  const __bf16* w2b = W2 + (size_t)(col0 + nlane) * I_DIM;

  const v8f vzero = {0.f, 0.f, 0.f, 0.f, 0.f, 0.f, 0.f, 0.f};
  v8f acca[4], accb[4];
#pragma unroll
  for (int t = 0; t < 4; ++t) { acca[t] = vzero; accb[t] = vzero; }

#pragma unroll 1
  for (int k0 = 0; k0 < I_DIM; k0 += 32) {
    const v16bf a0 = fragA(arow0, k0, lane);
    const v16bf a1 = fragA(arow1, k0, lane);
#pragma unroll
    for (int t = 0; t < 4; ++t) {
      const v16bf b = fragB(w2b + (size_t)t * 16 * I_DIM, k0, lane);
      acca[t] = WMMA_BF16(a0, b, acca[t]);
      accb[t] = WMMA_BF16(a1, b, accb[t]);
    }
  }

#pragma unroll
  for (int s = 0; s < 2; ++s) {
    const v8f* accs = s ? accb : acca;
#pragma unroll
    for (int g = 0; g < 8; ++g) {
      const int rowIdx = row0 + s * 16 + g + ((lane & 16) ? 8 : 0);
      if (rowIdx >= nTok) continue;
      const int tok = list ? list[rowIdx] : rowIdx;
      const float sc = scale[tok];
      float* orow = out + (size_t)tok * C_DIM + col0 + nlane;
#pragma unroll
      for (int t = 0; t < 4; ++t) {
        const float y = sc * accs[t][g];
        if (accumulate) orow[t * 16] += y;
        else            orow[t * 16] = y;
      }
    }
  }
}

// ---------------------------------------------------------------------------
// Host launcher
// ---------------------------------------------------------------------------
extern "C" void kernel_launch(void* const* d_in, const int* in_sizes, int n_in,
                              void* d_out, int out_size, void* d_ws,
                              size_t ws_size, hipStream_t stream) {
  (void)in_sizes; (void)n_in; (void)out_size; (void)ws_size;

  const float* x    = (const float*)d_in[0];
  const float* rw   = (const float*)d_in[1];
  const float* rb   = (const float*)d_in[2];
  const float* sw1  = (const float*)d_in[3];
  const float* sw2  = (const float*)d_in[4];
  const float* sw3  = (const float*)d_in[5];
  const float* ew1  = (const float*)d_in[6];
  const float* ew2  = (const float*)d_in[7];
  const float* ew3  = (const float*)d_in[8];
  float* out = (float*)d_out;

  // Workspace carve-up (256B aligned slices)
  char* ws = (char*)d_ws;
  size_t off = 0;
  auto carve = [&](size_t bytes) -> void* {
    void* p = (void*)(ws + off);
    off += (bytes + 255) & ~(size_t)255;
    return p;
  };
  const size_t IC = (size_t)I_DIM * C_DIM;
  float*  scale_sh = (float*)carve((size_t)NTOK * 4);
  float*  scale_sw = (float*)carve((size_t)NTOK * 4);
  int*    cnt      = (int*)carve((size_t)E_NUM * 4);
  int*    list     = (int*)carve((size_t)E_NUM * NTOK * 4);
  __bf16* x_bf     = (__bf16*)carve((size_t)NTOK * C_DIM * 2);
  __bf16* sw1_bf   = (__bf16*)carve(IC * 2);
  __bf16* sw2_bf   = (__bf16*)carve(IC * 2);
  __bf16* sw3_bf   = (__bf16*)carve(IC * 2);
  __bf16* ew1_bf   = (__bf16*)carve((size_t)E_NUM * IC * 2);
  __bf16* ew2_bf   = (__bf16*)carve((size_t)E_NUM * IC * 2);
  __bf16* ew3_bf   = (__bf16*)carve((size_t)E_NUM * IC * 2);
  __bf16* actS     = (__bf16*)carve((size_t)NTOK * I_DIM * 2);
  __bf16* actR     = (__bf16*)carve((size_t)NTOK * I_DIM * 2);

  hipMemsetAsync(cnt, 0, E_NUM * sizeof(int), stream);

  // 0) One-shot f32 -> bf16 materialization (bandwidth bound)
  const int CB = 256, CG = 4096;
  cvt_bf16_kernel<<<CG, CB, 0, stream>>>(x,   x_bf,   (long)NTOK * C_DIM);
  cvt_bf16_kernel<<<CG, CB, 0, stream>>>(sw1, sw1_bf, (long)IC);
  cvt_bf16_kernel<<<CG, CB, 0, stream>>>(sw2, sw2_bf, (long)IC);
  cvt_bf16_kernel<<<CG, CB, 0, stream>>>(sw3, sw3_bf, (long)IC);
  cvt_bf16_kernel<<<CG, CB, 0, stream>>>(ew1, ew1_bf, (long)E_NUM * IC);
  cvt_bf16_kernel<<<CG, CB, 0, stream>>>(ew2, ew2_bf, (long)E_NUM * IC);
  cvt_bf16_kernel<<<CG, CB, 0, stream>>>(ew3, ew3_bf, (long)E_NUM * IC);

  // 1) Router / routing scales / expert token lists
  moe_router_kernel<<<NTOK / 8, 256, 0, stream>>>(x, rw, rb, scale_sh,
                                                  scale_sw, cnt, list);

  // 2) Shared expert: up (fused SwiGLU gate) then down (writes out)
  {
    dim3 g(I_DIM / 256, NTOK / 32, 1);
    moe_up_kernel<<<g, 128, 0, stream>>>(x_bf, sw1_bf, sw3_bf, actS,
                                         nullptr, nullptr);
  }
  {
    dim3 g(C_DIM / 256, NTOK / 32, 1);
    moe_down_kernel<<<g, 128, 0, stream>>>(actS, sw2_bf, out, nullptr, nullptr,
                                           scale_sh, /*accumulate=*/0);
  }

  // 3) Switched experts: gather by routing, up then down (accumulates into out)
  {
    dim3 g(I_DIM / 256, NTOK / 32, E_NUM);
    moe_up_kernel<<<g, 128, 0, stream>>>(x_bf, ew1_bf, ew3_bf, actR, list, cnt);
  }
  {
    dim3 g(C_DIM / 256, NTOK / 32, E_NUM);
    moe_down_kernel<<<g, 128, 0, stream>>>(actR, ew2_bf, out, list, cnt,
                                           scale_sw, /*accumulate=*/1);
  }
}
